// GraphGenGnn_63780264345607
// MI455X (gfx1250) — matrix operand order repaired and checked by reading, not compiled, not run on hardware
//
#include <hip/hip_runtime.h>
#include <hip/hip_bf16.h>

typedef __attribute__((ext_vector_type(16))) _Float16 v16h;
typedef __attribute__((ext_vector_type(8)))  _Float16 v8h;
typedef __attribute__((ext_vector_type(8)))  float    v8f;

// ---------------------------------------------------------------------------
// Weight transpose + f32 -> f16 conversion.  W is (K x Nact) row-major f32.
// WT is (Npad x Kpad) row-major f16, zero padded.  total = Npad * Kpad.
// ---------------------------------------------------------------------------
__global__ void k_transpose_f16(const float* __restrict__ W, _Float16* __restrict__ WT,
                                int K, int Nact, int Kpad, int total) {
  int i = blockIdx.x * 256 + threadIdx.x;
  if (i >= total) return;
  int n = i / Kpad;
  int k = i - n * Kpad;
  float v = (k < K && n < Nact) ? W[(size_t)k * Nact + n] : 0.0f;
  WT[i] = (_Float16)v;
}

// ---------------------------------------------------------------------------
// Core WMMA GEMM:  C[M x Nout] = act( A[M x K] * B[K x Nout] + bias (+ef rows) )
// Block = 256 threads = 8 waves, BM=128 rows, BN=64 cols; each wave owns one
// 16-row strip and 4 column tiles (4 x v8f accumulators), so each A-fragment
// LDS read feeds 4 v_wmma and each barrier covers 32 wmma.
// Double-buffered LDS staging: next 128x32 A tile is loaded to registers
// before the WMMAs, stored to the alternate buffer after, one barrier/k-step.
// Block-uniform full-tile fast path removes exec-mask guards for interior
// blocks (all but the boundary block and the enc0 K-tail).
//  - BT is B^T: (Npad x Kpad) f16, Npad multiple of 64 so all B loads in-bounds.
//    Lane's B fragment = one contiguous 32B v16h load (B layout: n=L%16,
//    K = 16*(L/16)+h).
//  - A fragment: two contiguous 8-half LDS reads (A k order used here: halves
//    0-7 = K b..b+7, 8-15 = K b+8..b+15, b=16*(lane>=16)); stride 40 halves
//    makes the b128 reads bank-conflict free.
//  - ASRC 0: plain row-major f32 A (row stride Astride, valid K < Kact)
//    ASRC 1: A row m = X[idx[2m]] - X[idx[2m+1]], X row stride 256 (diff fuse)
//  - ATTADD: add BT[n][256+a0] + BT[n][261+a1] (two-hot edge-feature rows)
//  - ACT: 0 none, 1 relu, 2 sigmoid
// ---------------------------------------------------------------------------
template<int ASRC, int ACT, int ATTADD>
__global__ __launch_bounds__(256)
void gemm16(const float* __restrict__ A, const int* __restrict__ pidx,
            const int* __restrict__ eatt,
            const _Float16* __restrict__ BT, const float* __restrict__ bias,
            float* __restrict__ C,
            int M, int Nout, int Kloop, int Kact, int Kpad, int Astride) {
  __shared__ _Float16 As[2][128 * 40];        // double buffer, stride 40 halves

  const int mBase = blockIdx.x * 128;
  const int n0    = blockIdx.y * 64;
  const int t     = threadIdx.x;
  const int wv    = t >> 5;                   // wave id 0..7 -> 16-row strip
  const int lane  = t & 31;
  const int lhi   = lane >> 4;                // 0: lanes 0-15, 1: lanes 16-31
  const int l15   = lane & 15;
  const _Float16* brow = BT + (size_t)(n0 + l15) * Kpad + lhi * 16;
  const size_t bstep = (size_t)16 * Kpad;     // next 16-column tile
  const bool fullM = (mBase + 128 <= M);
  const int kq = t & 7;                       // 4-float chunk within 32-k tile
  const int r0 = t >> 3;                      // 32 rows per pass

  float rv[4][4];                             // staged A tile (regs)

  auto loadTile = [&](int kk) {
    const int kg = kk + kq * 4;
    const bool safeK = (ASRC != 0) || (kg + 3 < Kact);
    if (fullM && safeK) {                     // fast path: no guards
#pragma unroll
      for (int p = 0; p < 4; ++p) {
        const int m = mBase + r0 + p * 32;
        if (ASRC == 0) {
          const float* ap = A + (size_t)m * Astride + kg;
          rv[p][0] = ap[0]; rv[p][1] = ap[1]; rv[p][2] = ap[2]; rv[p][3] = ap[3];
        } else {
          const int i0 = pidx[2 * m], i1 = pidx[2 * m + 1];
          const float* p0 = A + (size_t)i0 * 256 + kg;
          const float* p1 = A + (size_t)i1 * 256 + kg;
          rv[p][0] = p0[0] - p1[0]; rv[p][1] = p0[1] - p1[1];
          rv[p][2] = p0[2] - p1[2]; rv[p][3] = p0[3] - p1[3];
        }
      }
    } else {                                  // boundary blocks / K tail
#pragma unroll
      for (int p = 0; p < 4; ++p) {
        const int m = mBase + r0 + p * 32;
        rv[p][0] = 0.f; rv[p][1] = 0.f; rv[p][2] = 0.f; rv[p][3] = 0.f;
        if (m < M) {
          if (ASRC == 0) {
            const float* ap = A + (size_t)m * Astride;
            if (kg     < Kact) rv[p][0] = ap[kg];
            if (kg + 1 < Kact) rv[p][1] = ap[kg + 1];
            if (kg + 2 < Kact) rv[p][2] = ap[kg + 2];
            if (kg + 3 < Kact) rv[p][3] = ap[kg + 3];
          } else {
            const int i0 = pidx[2 * m], i1 = pidx[2 * m + 1];
            const float* p0 = A + (size_t)i0 * 256 + kg;
            const float* p1 = A + (size_t)i1 * 256 + kg;
            rv[p][0] = p0[0] - p1[0]; rv[p][1] = p0[1] - p1[1];
            rv[p][2] = p0[2] - p1[2]; rv[p][3] = p0[3] - p1[3];
          }
        }
      }
    }
  };
  auto storeTile = [&](int buf) {
#pragma unroll
    for (int p = 0; p < 4; ++p) {
      _Float16* dst = &As[buf][(r0 + p * 32) * 40 + kq * 4];
      dst[0] = (_Float16)rv[p][0]; dst[1] = (_Float16)rv[p][1];
      dst[2] = (_Float16)rv[p][2]; dst[3] = (_Float16)rv[p][3];
    }
  };

  v8f acc[4] = {};

  loadTile(0);
  storeTile(0);
  __syncthreads();

  for (int kk = 0; kk < Kloop; kk += 32) {
    const int buf = (kk >> 5) & 1;
    const bool more = (kk + 32) < Kloop;
    if (more) {
      loadTile(kk + 32);                      // overlap with WMMAs below
      __builtin_prefetch(brow + kk + 32, 0, 1);              // global_prefetch_b8
      __builtin_prefetch(brow + 2 * bstep + kk + 32, 0, 1);
    }

    // ---- A fragment once, 4 B fragments + 4 WMMAs ----
    const _Float16* arow = &As[buf][(wv * 16 + l15) * 40 + lhi * 16];
    v8h alo = *(const v8h*)(arow);            // K = 16*lhi + 0..7
    v8h ahi = *(const v8h*)(arow + 8);        // K = 16*lhi + 8..15
    v16h a;
#pragma unroll
    for (int i = 0; i < 8; ++i) { a[i] = alo[i]; a[8 + i] = ahi[i]; }
#pragma unroll
    for (int nt = 0; nt < 4; ++nt) {
      v16h b = *(const v16h*)(brow + nt * bstep + kk);       // K = 16*lhi + 0..15
      acc[nt] = __builtin_amdgcn_wmma_f32_16x16x32_f16(
          false, a, false, b, (short)0, acc[nt], false, false);
    }

    if (more) storeTile(buf ^ 1);
    __syncthreads();
  }

  // ---- epilogue: bias, optional edge-feature rows, activation, store ----
  float bv[4];
#pragma unroll
  for (int nt = 0; nt < 4; ++nt) {
    const int n = n0 + nt * 16 + l15;
    bv[nt] = (n < Nout) ? bias[n] : 0.f;
  }
#pragma unroll
  for (int v = 0; v < 8; ++v) {
    const int m = mBase + wv * 16 + lhi * 8 + v;   // C layout: VGPR v -> M=8*lhi+v
    int e = 0;
    if (ATTADD) { if (m < M) e = eatt[m]; }
#pragma unroll
    for (int nt = 0; nt < 4; ++nt) {
      const int n = n0 + nt * 16 + l15;
      float x = acc[nt][v] + bv[nt];
      if (ATTADD) {
        if (m < M) {
          x += (float)BT[(size_t)n * Kpad + 256 + (e & 0xFF)]
             + (float)BT[(size_t)n * Kpad + 261 + ((e >> 8) & 0xFF)];
        }
      }
      if (ACT == 1) x = fmaxf(x, 0.f);
      if (ACT == 2) x = 1.f / (1.f + __expf(-x));
      if (m < M && n < Nout) C[(size_t)m * Nout + n] = x;
    }
  }
}

// ---------------------------------------------------------------------------
// Small elementwise / gather / scatter kernels
// ---------------------------------------------------------------------------
__global__ void k_gather_state(const float* __restrict__ nf, const int* __restrict__ idx,
                               float* __restrict__ st, int total, int NFEAT) {
  int i = blockIdx.x * 256 + threadIdx.x;
  if (i >= total) return;
  int nrow = i >> 8, c = i & 255;
  int j = idx[nrow];
  st[i] = (j < NFEAT) ? nf[(size_t)j * 256 + c] : 0.f;   // row NFEAT == zero row
}

__global__ void k_build_eatt(const int* __restrict__ edges, const int* __restrict__ att,
                             int* __restrict__ eatt, int M) {
  int m = blockIdx.x * 256 + threadIdx.x;
  if (m < M)
    eatt[m] = (att[edges[2 * m]] & 0xFF) | ((att[edges[2 * m + 1]] & 0xFF) << 8);
}

__global__ void k_relu(float* __restrict__ x, int total) {
  int i = blockIdx.x * 256 + threadIdx.x;
  if (i < total) x[i] = fmaxf(x[i], 0.f);
}

__global__ void k_zero(float* __restrict__ x, int total) {
  int i = blockIdx.x * 256 + threadIdx.x;
  if (i < total) x[i] = 0.f;
}

// agg[e1[m]] += msg[m]*att_gate[m]  (segment_sum over destination node)
__global__ void k_gate_scatter(const float* __restrict__ msg, const float* __restrict__ att,
                               const int* __restrict__ edges, float* __restrict__ agg,
                               int total) {
  int i = blockIdx.x * 256 + threadIdx.x;
  if (i >= total) return;
  int m = i >> 7, c = i & 127;
  int dst = edges[2 * m + 1];
  atomicAdd(&agg[(size_t)dst * 128 + c], msg[i] * att[i]);
}

// GRU gates: gi/gh are (N x 768) with [r|z|n] blocks of 256; h,out are (N x 256)
__global__ void k_gru(const float* __restrict__ gi, const float* __restrict__ gh,
                      const float* __restrict__ h, float* __restrict__ out, int total) {
  int i = blockIdx.x * 256 + threadIdx.x;
  if (i >= total) return;
  int nrow = i >> 8, c = i & 255;
  size_t b = (size_t)nrow * 768;
  float r  = 1.f / (1.f + __expf(-(gi[b + c]       + gh[b + c])));
  float z  = 1.f / (1.f + __expf(-(gi[b + 256 + c] + gh[b + 256 + c])));
  float tn = tanhf(gi[b + 512 + c] + r * gh[b + 512 + c]);
  out[i] = (1.f - z) * tn + z * h[i];
}

// ---------------------------------------------------------------------------
// Host launcher
// ---------------------------------------------------------------------------
extern "C" void kernel_launch(void* const* d_in, const int* in_sizes, int n_in,
                              void* d_out, int out_size, void* d_ws, size_t ws_size,
                              hipStream_t stream) {
  (void)n_in; (void)out_size; (void)ws_size;
  const float* A_pad    = (const float*)d_in[0];
  const int*   node_idx = (const int*)d_in[1];
  const int*   edges    = (const int*)d_in[2];
  const int*   att_idx  = (const int*)d_in[3];
  const int*   pairs    = (const int*)d_in[4];
  const int NFEAT = in_sizes[0] / 1000;       // 2*MAX_N = 1000
  const int Nn    = in_sizes[1];
  const int M     = in_sizes[2] / 2;
  const int P     = in_sizes[4] / 2;
  float* out = (float*)d_out;

  // params flattened in jax tree order (dict keys sorted, lists in order):
  // alpha[5..10], enc[11..16], layers l: base=17+12l
  //   {att b0,w0,b1,w1 | gru b_hh,b_ih,w_hh,w_ih | msg b0,w0,b1,w1}, theta[101..106]
  auto L = [&](int i) -> const float* { return (const float*)d_in[i]; };

  // ---- workspace carving ----
  size_t off = 0;
  auto allocB = [&](size_t bytes) -> void* {
    off = (off + 255) & ~(size_t)255;
    void* p = (char*)d_ws + off;
    off += bytes;
    return p;
  };
  auto allocH = [&](size_t n) -> _Float16* { return (_Float16*)allocB(n * 2); };
  auto allocF = [&](size_t n) -> float*    { return (float*)allocB(n * 4); };
  auto allocI = [&](size_t n) -> int*      { return (int*)allocB(n * 4); };

  _Float16* wt_enc0 = allocH(128 * 1024);
  _Float16* wt_enc1 = allocH(128 * 128);
  _Float16* wt_enc2 = allocH(256 * 128);
  _Float16 *wt_m0[7], *wt_m1[7], *wt_a0[7], *wt_a1[7], *wt_ih[7], *wt_hh[7];
  for (int l = 0; l < 7; ++l) {
    wt_a0[l] = allocH(128 * 288);  wt_a1[l] = allocH(128 * 128);
    wt_ih[l] = allocH(768 * 128);  wt_hh[l] = allocH(768 * 256);
    wt_m0[l] = allocH(128 * 288);  wt_m1[l] = allocH(128 * 128);
  }
  _Float16* wt_t0 = allocH(128 * 256);
  _Float16* wt_t1 = allocH(128 * 128);
  _Float16* wt_t2 = allocH(64 * 128);    // Npad 64 (Nout 40)
  _Float16* wt_l0 = allocH(128 * 256);
  _Float16* wt_l1 = allocH(128 * 128);
  _Float16* wt_l2 = allocH(64 * 128);    // Npad 64 (Nout 20)

  float* nf   = allocF((size_t)NFEAT * 256);
  float* sA   = allocF((size_t)Nn * 256);
  float* sB   = allocF((size_t)Nn * 256);
  float* agg  = allocF((size_t)Nn * 128);
  int*   eatt = allocI((size_t)M);

  size_t af = (size_t)3 * M * 128;
  size_t gf = (size_t)2 * Nn * 768; if (gf > af) af = gf;
  size_t hf = (size_t)2 * P * 128;  if (hf > af) af = hf;
  float* arena = allocF(af);
  float* E1 = arena;                          // edge bufs (layer phase)
  float* E2 = arena + (size_t)M * 128;
  float* E3 = arena + (size_t)2 * M * 128;
  float* gi = arena;                          // GRU bufs (after scatter consumed E*)
  float* gh = arena + (size_t)Nn * 768;
  float* hA = arena;                          // head / encoder bufs
  float* hB = arena + (size_t)(P > NFEAT ? P : NFEAT) * 128;

  // ---- weight prep (transpose + f16 convert) ----
  auto xp = [&](const float* W, _Float16* WT, int K, int Nact, int Kpad, int Npad) {
    int total = Npad * Kpad;
    k_transpose_f16<<<(total + 255) / 256, 256, 0, stream>>>(W, WT, K, Nact, Kpad, total);
  };
  xp(L(12), wt_enc0, 1000, 128, 1024, 128);
  xp(L(14), wt_enc1, 128, 128, 128, 128);
  xp(L(16), wt_enc2, 128, 256, 128, 256);
  for (int l = 0; l < 7; ++l) {
    int Lb = 17 + 12 * l;
    xp(L(Lb + 1), wt_a0[l], 266, 128, 288, 128);
    xp(L(Lb + 3), wt_a1[l], 128, 128, 128, 128);
    xp(L(Lb + 6), wt_hh[l], 256, 768, 256, 768);
    xp(L(Lb + 7), wt_ih[l], 128, 768, 128, 768);
    xp(L(Lb + 9), wt_m0[l], 266, 128, 288, 128);
    xp(L(Lb + 11), wt_m1[l], 128, 128, 128, 128);
  }
  xp(L(102), wt_t0, 256, 128, 256, 128);
  xp(L(104), wt_t1, 128, 128, 128, 128);
  xp(L(106), wt_t2, 128, 40, 128, 64);
  xp(L(6),  wt_l0, 256, 128, 256, 128);
  xp(L(8),  wt_l1, 128, 128, 128, 128);
  xp(L(10), wt_l2, 128, 20, 128, 64);

  // ---- GEMM dispatcher: BM=128, BN=64 ----
  auto gemm = [&](int asrc, int act, int attadd,
                  const float* Aa, const int* pidx, const int* ea,
                  const _Float16* BT, const float* bias, float* Cc,
                  int Mm, int Nout, int Kloop, int Kact, int Kpad, int Astride) {
    dim3 g((Mm + 127) / 128, (Nout + 63) / 64), b(256);
    if (asrc == 1 && attadd)
      gemm16<1, 1, 1><<<g, b, 0, stream>>>(Aa, pidx, ea, BT, bias, Cc, Mm, Nout, Kloop, Kact, Kpad, Astride);
    else if (asrc == 1)
      gemm16<1, 1, 0><<<g, b, 0, stream>>>(Aa, pidx, ea, BT, bias, Cc, Mm, Nout, Kloop, Kact, Kpad, Astride);
    else if (act == 1)
      gemm16<0, 1, 0><<<g, b, 0, stream>>>(Aa, pidx, ea, BT, bias, Cc, Mm, Nout, Kloop, Kact, Kpad, Astride);
    else if (act == 2)
      gemm16<0, 2, 0><<<g, b, 0, stream>>>(Aa, pidx, ea, BT, bias, Cc, Mm, Nout, Kloop, Kact, Kpad, Astride);
    else
      gemm16<0, 0, 0><<<g, b, 0, stream>>>(Aa, pidx, ea, BT, bias, Cc, Mm, Nout, Kloop, Kact, Kpad, Astride);
  };

  // ---- encoder: A_pad -> nf (NFEAT x 256) ----
  gemm(0, 1, 0, A_pad, nullptr, nullptr, wt_enc0, L(11), hA, NFEAT, 128, 1024, 1000, 1024, 1000);
  gemm(0, 1, 0, hA,    nullptr, nullptr, wt_enc1, L(13), hB, NFEAT, 128, 128, 128, 128, 128);
  gemm(0, 0, 0, hB,    nullptr, nullptr, wt_enc2, L(15), nf, NFEAT, 256, 128, 128, 128, 128);
  k_gather_state<<<(Nn * 256 + 255) / 256, 256, 0, stream>>>(nf, node_idx, sA, Nn * 256, NFEAT);
  k_build_eatt<<<(M + 255) / 256, 256, 0, stream>>>(edges, att_idx, eatt, M);

  // ---- 7 message-passing layers ----
  float* scur = sA;
  float* snew = sB;
  for (int l = 0; l < 7; ++l) {
    int Lb = 17 + 12 * l;
    if (l > 0) k_relu<<<(Nn * 256 + 255) / 256, 256, 0, stream>>>(scur, Nn * 256);
    // att branch: relu(ein@Wa0 + ef-rows + b) -> sigmoid(h@Wa1 + b)
    gemm(1, 1, 1, scur, edges, eatt, wt_a0[l], L(Lb + 0), E1, M, 128, 256, 256, 288, 0);
    gemm(0, 2, 0, E1, nullptr, nullptr, wt_a1[l], L(Lb + 2), E2, M, 128, 128, 128, 128, 128);
    // msg branch: relu(...) -> linear
    gemm(1, 1, 1, scur, edges, eatt, wt_m0[l], L(Lb + 8), E1, M, 128, 256, 256, 288, 0);
    gemm(0, 0, 0, E1, nullptr, nullptr, wt_m1[l], L(Lb + 10), E3, M, 128, 128, 128, 128, 128);
    // segment-sum of gated messages
    k_zero<<<(Nn * 128 + 255) / 256, 256, 0, stream>>>(agg, Nn * 128);
    k_gate_scatter<<<(M * 128 + 255) / 256, 256, 0, stream>>>(E3, E2, edges, agg, M * 128);
    // GRU: gi = agg@w_ih + b_ih ; gh = state@w_hh + b_hh ; gate fuse
    gemm(0, 0, 0, agg,  nullptr, nullptr, wt_ih[l], L(Lb + 5), gi, Nn, 768, 128, 128, 128, 128);
    gemm(0, 0, 0, scur, nullptr, nullptr, wt_hh[l], L(Lb + 4), gh, Nn, 768, 256, 256, 256, 256);
    k_gru<<<(Nn * 256 + 255) / 256, 256, 0, stream>>>(gi, gh, scur, snew, Nn * 256);
    float* tmp = scur; scur = snew; snew = tmp;
  }

  // ---- output heads on d = state[i0]-state[i1] ----
  gemm(1, 1, 0, scur, pairs, nullptr, wt_t0, L(101), hA, P, 128, 256, 256, 256, 0);
  gemm(0, 1, 0, hA, nullptr, nullptr, wt_t1, L(103), hB, P, 128, 128, 128, 128, 128);
  gemm(0, 0, 0, hB, nullptr, nullptr, wt_t2, L(105), out, P, 40, 128, 128, 128, 128);
  gemm(1, 1, 0, scur, pairs, nullptr, wt_l0, L(5), hA, P, 128, 256, 256, 256, 0);
  gemm(0, 1, 0, hA, nullptr, nullptr, wt_l1, L(7), hB, P, 128, 128, 128, 128, 128);
  gemm(0, 0, 0, hB, nullptr, nullptr, wt_l2, L(9), out + (size_t)P * 40, P, 20, 128, 128, 128, 128);
}
